// EssenceNetClassifier_49727131353479
// MI455X (gfx1250) — matrix operand order repaired and checked by reading, not compiled, not run on hardware
//
#include <hip/hip_runtime.h>

typedef __bf16 bf16;
typedef __attribute__((ext_vector_type(8)))  __bf16 bf16x8;
typedef __attribute__((ext_vector_type(16))) __bf16 v16bf;
typedef __attribute__((ext_vector_type(8)))  float  v8f;

#define DEVINL __device__ __forceinline__

DEVINL v16bf cat8(bf16x8 a, bf16x8 b) {
  return __builtin_shufflevector(a, b, 0, 1, 2, 3, 4, 5, 6, 7, 8, 9, 10, 11, 12, 13, 14, 15);
}
DEVINL v8f wmma_bf16(v16bf a, v16bf b, v8f c) {
  return __builtin_amdgcn_wmma_f32_16x16x32_bf16(false, a, false, b, (short)0, c, false, false);
}
// slow-path element K maps (contiguous k = 16*g + e, same for A 16x32 and B 32x16)
DEVINL int a_kidx(int e, int g) { return e + (g << 4); }
DEVINL int b_kidx(int e, int g) { return e + (g << 4); }

// ---------------- Tensor Data Mover (gfx1250) ----------------
#if defined(__has_builtin)
#if __has_builtin(__builtin_amdgcn_tensor_load_to_lds)
#define HAVE_TDM 1
#endif
#endif
#ifndef HAVE_TDM
#define HAVE_TDM 0
#endif

#if HAVE_TDM
typedef __attribute__((ext_vector_type(4))) unsigned int u32x4;
typedef __attribute__((ext_vector_type(4))) int i32x4;
typedef __attribute__((ext_vector_type(8))) int i32x8;
// 2D tile load: 112 rows x 32 bf16, row stride Kp elements, into LDS at lds_off.
DEVINL void tdm_load_tile(unsigned lds_off, unsigned long long gaddr, int Kp) {
  u32x4 g0;
  g0[0] = 1u;                                                   // count=1 (valid user D#)
  g0[1] = lds_off;                                              // lds_addr (bytes)
  g0[2] = (unsigned)gaddr;                                      // global_addr[31:0]
  g0[3] = ((unsigned)(gaddr >> 32) & 0x01FFFFFFu) | (2u << 30); // addr[56:32] | type=2
  i32x8 g1;
  g1[0] = (int)(1u << 16);                                      // data_size = 2 bytes
  g1[1] = (int)(((unsigned)Kp & 0xFFFFu) << 16);                // tensor_dim0[15:0]
  g1[2] = (int)(((unsigned)Kp >> 16) | (112u << 16));           // tensor_dim0[31:16] | tensor_dim1[15:0]
  g1[3] = (int)(32u << 16);                                     // tile_dim0 = 32
  g1[4] = 112;                                                  // tile_dim1 = 112, tile_dim2 = 0
  g1[5] = Kp;                                                   // tensor_dim0_stride[31:0]
  g1[6] = 0;
  g1[7] = 0;
  i32x4 z4 = {};
#if __has_include(<hip/amd_detail/amd_gfx1250_TDM.h>)
  i32x8 z8 = {};
  __builtin_amdgcn_tensor_load_to_lds(g0, g1, z4, z4, z8, 0);   // clang-23 6-arg form
#else
  __builtin_amdgcn_tensor_load_to_lds(g0, g1, z4, z4, 0);       // ROCm 7.2 5-arg form
#endif
}
DEVINL void tdm_wait() { __builtin_amdgcn_s_wait_tensorcnt(0); }
#endif

// ---------------- utility kernels ----------------

__global__ void k_zero_f(float* __restrict__ p, int n) {
  int i = blockIdx.x * blockDim.x + threadIdx.x;
  if (i < n) p[i] = 0.f;
}
__global__ void k_zero_u64(unsigned long long* __restrict__ p, int n) {
  int i = blockIdx.x * blockDim.x + threadIdx.x;
  if (i < n) p[i] = 0ull;
}

__global__ void k_gray(const float* __restrict__ x, bf16* __restrict__ gray) {
  int i = blockIdx.x * blockDim.x + threadIdx.x;  // 16*65536
  if (i >= 16 * 65536) return;
  int b = i >> 16, p = i & 65535;
  const float* xb = x + (size_t)b * 3 * 65536;
  gray[i] = (bf16)(0.299f * xb[p] + 0.587f * xb[65536 + p] + 0.114f * xb[131072 + p]);
}

__global__ void k_pool2(const float* __restrict__ src, float* __restrict__ dst, int Ho) {
  int i = blockIdx.x * blockDim.x + threadIdx.x;
  if (i >= 48 * Ho * Ho) return;
  int ow = i % Ho; int t = i / Ho; int oh = t % Ho; int bc = t / Ho;
  int Hi = Ho * 2;
  const float* s = src + ((size_t)bc * Hi + oh * 2) * Hi + ow * 2;
  dst[i] = 0.25f * (s[0] + s[1] + s[Hi] + s[Hi + 1]);
}

__global__ void k_rc(const float* __restrict__ pyr, bf16* __restrict__ cc,
                     int HH, int Kp, int chanOff, int rep) {
  int i = blockIdx.x * blockDim.x + threadIdx.x;
  if (i >= 16 * 3 * HH) return;
  int p = i % HH; int t = i / HH; int c = t % 3; int b = t / 3;
  bf16 v = (bf16)pyr[i];
  for (int r = 0; r < rep; ++r)
    cc[((size_t)b * HH * rep + p * rep + r) * Kp + chanOff + c] = v;
}

__global__ void k_packw(const float* __restrict__ W, bf16* __restrict__ out,
                        int N, int K, int Npad, int Kp) {
  int i = blockIdx.x * blockDim.x + threadIdx.x;
  if (i >= Npad * Kp) return;
  int k = i % Kp, n = i / Kp;
  out[i] = (bf16)((n < N && k < K) ? W[(size_t)n * K + k] : 0.f);
}

// ---------------- conv-as-GEMM (stride==kernel, levels 0..6) ----------------
template <bool FAST>
__global__ void k_conv_gemm(const bf16* __restrict__ gray, const float* __restrict__ W,
                            float* __restrict__ out, int lh, int lks, int OUTC,
                            int Ktot, int Kchunk) {
  int wave = threadIdx.x >> 5, lane = threadIdx.x & 31;
  int H = 1 << lh, HH = 1 << (lh + lh);
  int ntiles = OUTC >> 4;
  int tile = blockIdx.x * (blockDim.x >> 5) + wave;
  if (tile >= HH * ntiles) return;  // wave-uniform
  int mt = tile / ntiles, nt = tile % ntiles;
  int g = lane >> 4, l16 = lane & 15;
  int KS = 1 << lks;

  int m = (mt << 4) + l16;
  int b = m >> (lh + lh);
  int pix = m & (HH - 1);
  int oh = pix >> lh, ow = pix & (H - 1);
  const bf16* gb = gray + ((size_t)b << 16);
  int pbase = (oh << (lks + 8)) + (ow << lks);
  int n = (nt << 4) + l16;
  const float* wrow = W + (size_t)n * Ktot;

  int k0 = blockIdx.y * Kchunk;
  int k1 = k0 + Kchunk; if (k1 > Ktot) k1 = Ktot;

  v8f acc = {};
  for (int kk = k0; kk < k1; kk += 32) {
    __builtin_prefetch(wrow + kk + 128, 0, 1);
    v16bf a, bb;
    if (FAST) {
      int ka0 = kk + (g << 4), ka1 = ka0 + 8;
      bf16x8 r0 = *(const bf16x8*)(gb + pbase + ((ka0 >> lks) << 8) + (ka0 & (KS - 1)));
      bf16x8 r1 = *(const bf16x8*)(gb + pbase + ((ka1 >> lks) << 8) + (ka1 & (KS - 1)));
      const float* wp = wrow + kk + (g << 4);
      float4 w0 = *(const float4*)(wp + 0);
      float4 w1 = *(const float4*)(wp + 4);
      float4 w2 = *(const float4*)(wp + 8);
      float4 w3 = *(const float4*)(wp + 12);
      a = cat8(r0, r1);
      bb[0] = (bf16)w0.x; bb[1] = (bf16)w0.y; bb[2] = (bf16)w0.z; bb[3] = (bf16)w0.w;
      bb[4] = (bf16)w1.x; bb[5] = (bf16)w1.y; bb[6] = (bf16)w1.z; bb[7] = (bf16)w1.w;
      bb[8] = (bf16)w2.x; bb[9] = (bf16)w2.y; bb[10] = (bf16)w2.z; bb[11] = (bf16)w2.w;
      bb[12] = (bf16)w3.x; bb[13] = (bf16)w3.y; bb[14] = (bf16)w3.z; bb[15] = (bf16)w3.w;
    } else {
#pragma unroll
      for (int e = 0; e < 16; ++e) {  // clamp+select: no EXEC divergence
        int ka = kk + a_kidx(e, g);
        int kc = (ka < Ktot) ? ka : 0;
        float av = (float)gb[pbase + ((kc >> lks) << 8) + (kc & (KS - 1))];
        a[e] = (bf16)((ka < Ktot) ? av : 0.f);
        int kb = kk + b_kidx(e, g);
        float bv = wrow[(kb < Ktot) ? kb : 0];
        bb[e] = (bf16)((kb < Ktot) ? bv : 0.f);
      }
    }
    acc = wmma_bf16(a, bb, acc);
  }
#pragma unroll
  for (int v = 0; v < 8; ++v) {
    int mm = (mt << 4) + v + (g << 3);
    int b2 = mm >> (lh + lh);
    int pp = mm & (HH - 1);
    atomicAdd(&out[((size_t)b2 * OUTC + ((nt << 4) + l16)) * HH + pp], acc[v]);
  }
}

// level 7: 3x3 stride-2 pad-1 direct conv
__global__ void k_conv7(const bf16* __restrict__ gray, const float* __restrict__ W,
                        float* __restrict__ out) {
  int i = blockIdx.x * blockDim.x + threadIdx.x;
  if (i >= 16 * 8 * 128 * 128) return;
  int ow = i & 127; int t = i >> 7; int oh = t & 127; t >>= 7; int c = t & 7; int b = t >> 3;
  const bf16* gb = gray + ((size_t)b << 16);
  const float* w = W + c * 9;
  int iy0 = oh * 2 - 1, ix0 = ow * 2 - 1;
  float s = 0.f;
#pragma unroll
  for (int ky = 0; ky < 3; ++ky) {
    int iy = iy0 + ky;
    if ((unsigned)iy < 256u)
#pragma unroll
      for (int kx = 0; kx < 3; ++kx) {
        int ix = ix0 + kx;
        if ((unsigned)ix < 256u) s += w[ky * 3 + kx] * (float)gb[(iy << 8) + ix];
      }
  }
  out[i] = s;
}

// GroupNorm + SiLU -> cc channels [chanOff + c], optional x4 replicate (level 0)
__global__ void k_gn_silu(const float* __restrict__ src, const float* __restrict__ gw,
                          const float* __restrict__ gbias, bf16* __restrict__ cc,
                          int C, int G, int HH, int Kp, int chanOff, int rep) {
  int b = blockIdx.x / G, g = blockIdx.x % G;
  int cpg = C / G;
  int n = cpg * HH;
  const float* base = src + ((size_t)b * C + g * cpg) * HH;
  __shared__ float s1[256], s2[256];
  float sum = 0.f, sq = 0.f;
  for (int i = threadIdx.x; i < n; i += blockDim.x) { float v = base[i]; sum += v; sq += v * v; }
  s1[threadIdx.x] = sum; s2[threadIdx.x] = sq; __syncthreads();
  for (int o = 128; o > 0; o >>= 1) {
    if ((int)threadIdx.x < o) { s1[threadIdx.x] += s1[threadIdx.x + o]; s2[threadIdx.x] += s2[threadIdx.x + o]; }
    __syncthreads();
  }
  float mean = s1[0] / n;
  float rstd = rsqrtf(s2[0] / n - mean * mean + 1e-5f);
  for (int i = threadIdx.x; i < n; i += blockDim.x) {
    int c = g * cpg + i / HH; int p = i % HH;
    float v = (base[i] - mean) * rstd * gw[c] + gbias[c];
    bf16 y = (bf16)(v / (1.f + __expf(-v)));
    for (int r = 0; r < rep; ++r)
      cc[((size_t)b * HH * rep + p * rep + r) * Kp + chanOff + c] = y;
  }
}

// ---------------- head: TDM-staged B, WMMA, fused softmax-max + argmax ----------------
__global__ void k_head(const bf16* __restrict__ cc, const bf16* __restrict__ Wp,
                       const float* __restrict__ bias, unsigned long long* __restrict__ best,
                       int lvl, int Kp, int lh) {
  __shared__ bf16 smem[2][112 * 32];  // double-buffered 112x32 B tile (7KB each)
  int wave = threadIdx.x >> 5, lane = threadIdx.x & 31;
  int HH = 1 << (lh + lh);
  int mt = blockIdx.x * (blockDim.x >> 5) + wave;
  if (mt >= HH) mt = HH - 1;  // clamp (duplicates harmless) so all waves reach barriers
  int g = lane >> 4, l16 = lane & 15;
  const bf16* arow = cc + (size_t)((mt << 4) + l16) * Kp;

  v8f acc[7];
#pragma unroll
  for (int t = 0; t < 7; ++t) acc[t] = (v8f){};

#if HAVE_TDM
  unsigned long long wbase = (unsigned long long)(size_t)Wp;
  unsigned lds0 = (unsigned)(size_t)&smem[0][0];
  unsigned lds1 = (unsigned)(size_t)&smem[1][0];
  if (wave == 0) { tdm_load_tile(lds0, wbase, Kp); tdm_wait(); }
  __syncthreads();
  int cur = 0;
  for (int kk = 0; kk < Kp; kk += 32) {
    if (wave == 0 && kk + 32 < Kp)
      tdm_load_tile(cur ? lds0 : lds1, wbase + (size_t)(kk + 32) * 2, Kp);
    int ka0 = kk + (g << 4);
    v16bf a = cat8(*(const bf16x8*)(arow + ka0), *(const bf16x8*)(arow + ka0 + 8));
    v16bf bb[7];
#pragma unroll
    for (int t = 0; t < 7; ++t) {
      const bf16* sp = &smem[cur][((t << 4) + l16) * 32 + (g << 4)];
      bb[t] = cat8(*(const bf16x8*)sp, *(const bf16x8*)(sp + 8));
    }
#pragma unroll
    for (int t = 0; t < 7; ++t) acc[t] = wmma_bf16(a, bb[t], acc[t]);
    if (wave == 0) tdm_wait();
    __syncthreads();
    cur ^= 1;
  }
#else
  for (int kk = 0; kk < Kp; kk += 32) {
    int ka0 = kk + (g << 4);
    v16bf a = cat8(*(const bf16x8*)(arow + ka0), *(const bf16x8*)(arow + ka0 + 8));
    int kb = kk + (g << 4);
    v16bf bb[7];
#pragma unroll
    for (int t = 0; t < 7; ++t) {
      const bf16* wr = Wp + (size_t)((t << 4) + l16) * Kp + kb;
      bb[t] = cat8(*(const bf16x8*)(wr), *(const bf16x8*)(wr + 8));
    }
#pragma unroll
    for (int t = 0; t < 7; ++t) acc[t] = wmma_bf16(a, bb[t], acc[t]);
  }
#endif

#pragma unroll
  for (int t = 0; t < 7; ++t) {
    int n = (t << 4) + l16;
    float ba = (n < 100) ? bias[n] : -1e30f;
#pragma unroll
    for (int v = 0; v < 8; ++v) acc[t][v] = (n < 100) ? acc[t][v] + ba : -1e30f;
  }
#pragma unroll
  for (int v = 0; v < 8; ++v) {
    float mx = acc[0][v];
#pragma unroll
    for (int t = 1; t < 7; ++t) mx = fmaxf(mx, acc[t][v]);
    for (int o = 8; o >= 1; o >>= 1) mx = fmaxf(mx, __shfl_xor(mx, o, 16));
    float se = 0.f;
#pragma unroll
    for (int t = 0; t < 7; ++t) se += __expf(acc[t][v] - mx);
    for (int o = 8; o >= 1; o >>= 1) se += __shfl_xor(se, o, 16);
    if (l16 == 0) {
      float pmax = 1.f / se;
      int mm = (mt << 4) + v + (g << 3);
      int b2 = mm >> (lh + lh);
      int pp = mm & (HH - 1);
      unsigned long long key = ((unsigned long long)__float_as_uint(pmax) << 32) | (unsigned)pp;
      atomicMax(&best[lvl * 16 + b2], key);
    }
  }
}

// ---------------- projection: WMMA, writes x2 nearest-upsample into next cc ----------------
__global__ void k_proj(const bf16* __restrict__ cc, const bf16* __restrict__ Wp,
                       const float* __restrict__ bias, bf16* __restrict__ ccNext,
                       int Kp, int lh, int Fout, int KpNext) {
  int wave = threadIdx.x >> 5, lane = threadIdx.x & 31;
  int H = 1 << lh, HH = 1 << (lh + lh);
  int ntiles = (Fout + 15) >> 4;
  int tile = blockIdx.x * (blockDim.x >> 5) + wave;
  if (tile >= HH * ntiles) return;
  int mt = tile / ntiles, nt = tile % ntiles;
  int g = lane >> 4, l16 = lane & 15;
  const bf16* arow = cc + (size_t)((mt << 4) + l16) * Kp;
  int n = (nt << 4) + l16;
  const bf16* wrow = Wp + (size_t)n * Kp;

  v8f acc = {};
  for (int kk = 0; kk < Kp; kk += 32) {
    int ka0 = kk + (g << 4);
    int kb = kk + (g << 4);
    bf16x8 a0 = *(const bf16x8*)(arow + ka0);
    bf16x8 a1 = *(const bf16x8*)(arow + ka0 + 8);
    bf16x8 b0 = *(const bf16x8*)(wrow + kb);
    bf16x8 b1 = *(const bf16x8*)(wrow + kb + 8);
    acc = wmma_bf16(cat8(a0, a1), cat8(b0, b1), acc);
  }
  if (n < Fout) {
    float ba = bias[n];
    int H2 = H << 1;
#pragma unroll
    for (int v = 0; v < 8; ++v) {
      int mm = (mt << 4) + v + (g << 3);
      int b2 = mm >> (lh + lh);
      int pp = mm & (HH - 1);
      int y = pp >> lh, x = pp & (H - 1);
      bf16 val = (bf16)(acc[v] + ba);
      size_t base = ((size_t)b2 * (HH << 2) + (y << 1) * H2 + (x << 1)) * KpNext + n;
      ccNext[base] = val;
      ccNext[base + KpNext] = val;
      ccNext[base + (size_t)H2 * KpNext] = val;
      ccNext[base + (size_t)(H2 + 1) * KpNext] = val;
    }
  }
}

// ---------------- final selection ----------------
struct SelParams {
  const bf16* cc[7];
  const float* W[7];
  const float* bias[7];
  int Kp[7], lh[7], cin[7];
};

__global__ void k_select(SelParams P, const unsigned long long* __restrict__ best,
                         float* __restrict__ out) {
  int b = blockIdx.x;
  __shared__ int s_lvl, s_pix;
  __shared__ float red[128];
  if (threadIdx.x == 0) {
    int choice = 6;
    for (int l = 0; l < 7; ++l) {
      float pm = __uint_as_float((unsigned)(best[l * 16 + b] >> 32));
      if (pm > 0.8f) { choice = l; break; }
    }
    s_lvl = choice;
    s_pix = (int)(unsigned)best[choice * 16 + b];
  }
  __syncthreads();
  int l = s_lvl;
  int HH = 1 << (P.lh[l] + P.lh[l]);
  const bf16* row = P.cc[l] + (size_t)(b * HH + s_pix) * P.Kp[l];
  int cin = P.cin[l];
  int c = threadIdx.x;
  float logit = -1e30f;
  if (c < 100) {
    const float* w = P.W[l] + (size_t)c * cin;
    float s = P.bias[l][c];
    for (int k = 0; k < cin; ++k) s += (float)row[k] * w[k];
    logit = s;
  }
  red[threadIdx.x] = logit; __syncthreads();
  for (int o = 64; o > 0; o >>= 1) {
    if ((int)threadIdx.x < o) red[threadIdx.x] = fmaxf(red[threadIdx.x], red[threadIdx.x + o]);
    __syncthreads();
  }
  float mx = red[0]; __syncthreads();
  float e = (c < 100) ? __expf(logit - mx) : 0.f;
  red[threadIdx.x] = e; __syncthreads();
  for (int o = 64; o > 0; o >>= 1) {
    if ((int)threadIdx.x < o) red[threadIdx.x] += red[threadIdx.x + o];
    __syncthreads();
  }
  if (c < 100) out[b * 100 + c] = e / red[0];
}

// ---------------- host orchestration ----------------
extern "C" void kernel_launch(void* const* d_in, const int* in_sizes, int n_in,
                              void* d_out, int out_size, void* d_ws, size_t ws_size,
                              hipStream_t stream) {
  (void)in_sizes; (void)n_in; (void)out_size; (void)ws_size;
  const float* x = (const float*)d_in[0];
  const float *conv_w[8], *gn_w[8], *gn_b[8];
  const float *head_w[7], *head_b[7], *proj_w[7], *proj_b[7];
  for (int i = 0; i < 8; ++i) conv_w[i] = (const float*)d_in[1 + i];
  for (int i = 0; i < 8; ++i) gn_w[i] = (const float*)d_in[9 + i];
  for (int i = 0; i < 8; ++i) gn_b[i] = (const float*)d_in[17 + i];
  for (int i = 0; i < 7; ++i) head_w[i] = (const float*)d_in[25 + i];
  for (int i = 0; i < 7; ++i) head_b[i] = (const float*)d_in[32 + i];
  for (int i = 0; i < 7; ++i) proj_w[i] = (const float*)d_in[39 + i];
  for (int i = 0; i < 7; ++i) proj_b[i] = (const float*)d_in[46 + i];

  static const int OUTC[8] = {1024, 512, 256, 128, 64, 32, 16, 8};
  static const int GRP[8] = {32, 32, 16, 8, 4, 4, 4, 4};
  static const int FCH[8] = {1027, 515, 259, 131, 67, 35, 19, 11};
  int CIN[7], KP[7];
  for (int i = 0; i < 7; ++i) { CIN[i] = FCH[i] + FCH[i + 1]; KP[i] = (CIN[i] + 31) & ~31; }

  char* ws = (char*)d_ws;
  size_t off = 0;
  auto alloc = [&](size_t bytes) -> char* {
    char* p = ws + off;
    off += (bytes + 255) & ~(size_t)255;
    return p;
  };
  unsigned long long* best = (unsigned long long*)alloc(7 * 16 * 8);
  bf16* gray = (bf16*)alloc((size_t)16 * 65536 * 2);
  float* pyr[8];
  for (int l = 0; l < 8; ++l) pyr[l] = (float*)alloc((size_t)48 * (1u << (2 * l)) * 4);
  bf16* cc[7];
  size_t ccElems[7];
  for (int i = 0; i < 7; ++i) {
    ccElems[i] = (size_t)16 * (1u << (2 * (i + 1))) * KP[i];
    cc[i] = (bf16*)alloc(ccElems[i] * 2);
  }
  bf16* wh[7];
  for (int i = 0; i < 7; ++i) wh[i] = (bf16*)alloc((size_t)112 * KP[i] * 2);
  bf16* wp[6];
  int NPAD[6];
  for (int i = 0; i < 6; ++i) {
    NPAD[i] = (FCH[i + 1] + 15) & ~15;
    wp[i] = (bf16*)alloc((size_t)NPAD[i] * KP[i] * 2);
  }
  float* convout = (float*)alloc((size_t)16 * 8 * 16384 * 4);

  k_zero_u64<<<1, 128, 0, stream>>>(best, 7 * 16);
  for (int i = 0; i < 7; ++i) {
    int nf = (int)(ccElems[i] / 2);
    k_zero_f<<<(nf + 255) / 256, 256, 0, stream>>>((float*)cc[i], nf);
  }
  { int n = 16 * 65536; k_gray<<<(n + 255) / 256, 256, 0, stream>>>(x, gray); }
  { int Ho = 128; int n = 48 * Ho * Ho;
    k_pool2<<<(n + 255) / 256, 256, 0, stream>>>(x, pyr[7], Ho); }
  for (int l = 7; l >= 1; --l) {
    int Ho = 1 << (l - 1); int n = 48 * Ho * Ho;
    k_pool2<<<(n + 255) / 256, 256, 0, stream>>>(pyr[l], pyr[l - 1], Ho);
  }
  k_rc<<<1, 64, 0, stream>>>(pyr[0], cc[0], 1, KP[0], 0, 4);
  for (int l = 1; l <= 7; ++l) {
    int HH = 1 << (2 * l); int n = 16 * 3 * HH;
    k_rc<<<(n + 255) / 256, 256, 0, stream>>>(pyr[l], cc[l - 1], HH, KP[l - 1], FCH[l - 1], 1);
  }
  for (int i = 0; i < 7; ++i) {
    int n = 112 * KP[i];
    k_packw<<<(n + 255) / 256, 256, 0, stream>>>(head_w[i], wh[i], 100, CIN[i], 112, KP[i]);
  }
  for (int i = 0; i < 6; ++i) {
    int n = NPAD[i] * KP[i];
    k_packw<<<(n + 255) / 256, 256, 0, stream>>>(proj_w[i], wp[i], FCH[i + 1], CIN[i], NPAD[i], KP[i]);
  }

  static const int SPLITK[7] = {32, 16, 8, 4, 2, 1, 1};
  for (int l = 0; l < 7; ++l) {
    int lh = l, lks = 8 - l;
    int HH = 1 << (2 * l);
    int Ktot = 1 << (2 * lks);
    int n = 16 * OUTC[l] * HH;
    k_zero_f<<<(n + 255) / 256, 256, 0, stream>>>(convout, n);
    int tiles = HH * (OUTC[l] >> 4);
    int sk = SPLITK[l];
    int Kchunk = ((Ktot + sk - 1) / sk + 31) & ~31;
    dim3 grid((tiles + 7) / 8, sk);
    if (l < 6)
      k_conv_gemm<true><<<grid, 256, 0, stream>>>(gray, conv_w[l], convout, lh, lks, OUTC[l], Ktot, Kchunk);
    else
      k_conv_gemm<false><<<grid, 256, 0, stream>>>(gray, conv_w[l], convout, lh, lks, OUTC[l], Ktot, Kchunk);
    if (l == 0)
      k_gn_silu<<<16 * GRP[0], 256, 0, stream>>>(convout, gn_w[0], gn_b[0], cc[0],
                                                 OUTC[0], GRP[0], 1, KP[0], 3, 4);
    else
      k_gn_silu<<<16 * GRP[l], 256, 0, stream>>>(convout, gn_w[l], gn_b[l], cc[l - 1],
                                                 OUTC[l], GRP[l], HH, KP[l - 1], FCH[l - 1] + 3, 1);
  }
  { int n = 16 * 8 * 128 * 128;
    k_conv7<<<(n + 255) / 256, 256, 0, stream>>>(gray, conv_w[7], convout);
    k_gn_silu<<<16 * 4, 256, 0, stream>>>(convout, gn_w[7], gn_b[7], cc[6],
                                          8, 4, 16384, KP[6], FCH[6] + 3, 1); }

  SelParams SP;
  for (int i = 0; i < 7; ++i) {
    int lh = i + 1;
    int HH = 1 << (2 * lh);
    k_head<<<dim3((HH + 7) / 8), 256, 0, stream>>>(cc[i], wh[i], head_b[i], best, i, KP[i], lh);
    if (i < 6) {
      int tiles = HH * ((FCH[i + 1] + 15) >> 4);
      k_proj<<<dim3((tiles + 7) / 8), 256, 0, stream>>>(cc[i], wp[i], proj_b[i], cc[i + 1],
                                                        KP[i], lh, FCH[i + 1], KP[i + 1]);
    }
    SP.cc[i] = cc[i]; SP.W[i] = head_w[i]; SP.bias[i] = head_b[i];
    SP.Kp[i] = KP[i]; SP.lh[i] = lh; SP.cin[i] = CIN[i];
  }
  k_select<<<16, 128, 0, stream>>>(SP, best, (float*)d_out);
}
